// SparseTransFusionHead_88613765251662
// MI455X (gfx1250) — matrix operand order repaired and checked by reading, not compiled, not run on hardware
//
#include <hip/hip_runtime.h>

// ---------------- problem constants ----------------
constexpr int Bsz  = 2;
constexpr int CIN  = 256;
constexpr int C    = 128;
constexpr int H    = 180;
constexpr int W    = 180;
constexpr int HW   = H * W;      // 32400
constexpr int NCLS = 10;
constexpr int Kq   = 200;
constexpr int NH   = 8;
constexpr int DH   = 16;         // C / NH
constexpr int FFN  = 256;

typedef __attribute__((ext_vector_type(16))) __bf16 v16bf;
typedef __attribute__((ext_vector_type(8)))  __bf16 v8bf;
typedef __attribute__((ext_vector_type(8)))  float  v8f;

__device__ __forceinline__ __bf16 tobf(float f) { return (__bf16)f; }

__device__ __forceinline__ v8bf zero8() {
  v8bf z;
#pragma unroll
  for (int i = 0; i < 8; ++i) z[i] = (__bf16)0.f;
  return z;
}
__device__ __forceinline__ v8bf ld8(const __bf16* p) {       // 16B aligned
  return *(const v8bf*)p;
}
__device__ __forceinline__ void st8(__bf16* p, v8bf v) { *(v8bf*)p = v; }
__device__ __forceinline__ v16bf cat16(v8bf lo, v8bf hi) {
  return __builtin_shufflevector(lo, hi, 0, 1, 2, 3, 4, 5, 6, 7,
                                 8, 9, 10, 11, 12, 13, 14, 15);
}
// A-fragment of a K=32 tile = one contiguous 16-elem run at k = khalf*16 (same as B)
// B-fragment = one contiguous 16-elem run at k = khalf*16.

// =====================================================================
// 3x3 conv as implicit GEMM with bf16 WMMA.  out = relu?(acc*scale+shift)
// in: [B, cin, H, W] f32   wgt: OIHW [128, cin, 3, 3] f32   out: [B,128,H,W] f32
// Block: 128 threads (4 waves), each wave: 16 pixels x 16 out-channels.
// grid: (ceil(W/64), H, B * 8)
// =====================================================================
__global__ void conv3x3_wmma(const float* __restrict__ in,
                             const float* __restrict__ wgt,
                             const float* __restrict__ scale,
                             const float* __restrict__ shift,
                             int cin, int relu, float* __restrict__ out)
{
  __shared__ __align__(16) __bf16 lin[3][66][32];   // [dy][x][ci] ci-contiguous
  __shared__ __align__(16) __bf16 lw[9][16][32];    // [tap][co][ci]
  const int tid  = threadIdx.x;
  const int lane = tid & 31;
  const int wv   = tid >> 5;
  const int x0   = blockIdx.x * 64;
  const int y    = blockIdx.y;
  const int b    = blockIdx.z >> 3;
  const int n0   = (blockIdx.z & 7) * 16;

  v8f acc;
#pragma unroll
  for (int i = 0; i < 8; ++i) acc[i] = 0.f;

  const int m = lane & 15, kh = lane >> 4, n = lane & 15;

  for (int cb = 0; cb < cin; cb += 32) {
    // stage input rows y-1..y+1, 32 channels, x0-1 .. x0+64 (66 cols)
    for (int i = tid; i < 3 * 32 * 66; i += 128) {
      int dy = i / (32 * 66); int rem = i % (32 * 66);
      int ci = rem / 66;      int xx  = rem % 66;
      int gy = y + dy - 1;    int gx  = x0 + xx - 1;
      float v = 0.f;
      if (gy >= 0 && gy < H && gx >= 0 && gx < W)
        v = in[(((long)b * cin + cb + ci) * H + gy) * W + gx];
      lin[dy][xx][ci] = tobf(v);
    }
    // stage weights (ci-contiguous per [tap][co])
    for (int i = tid; i < 32 * 16 * 9; i += 128) {
      int k = i / (16 * 9); int rem = i % (16 * 9);
      int nn = rem / 9;     int t   = rem % 9;
      lw[t][nn][k] = tobf(wgt[((long)(n0 + nn) * cin + cb + k) * 9 + t]);
    }
    if (cb + 32 < cin)
      __builtin_prefetch(&in[(((long)b * cin + cb + 32) * H + y) * W + x0], 0, 1);
    __syncthreads();

#pragma unroll
    for (int tap = 0; tap < 9; ++tap) {
      int dy = tap / 3, dx = tap % 3;
      const __bf16* ar = &lin[dy][wv * 16 + m + dx][0];
      v16bf a  = cat16(ld8(ar + kh * 16), ld8(ar + kh * 16 + 8));
      const __bf16* br = &lw[tap][n][kh * 16];
      v16bf bf = cat16(ld8(br), ld8(br + 8));
      acc = __builtin_amdgcn_wmma_f32_16x16x32_bf16(false, a, false, bf,
                                                    (short)0, acc, false, false);
    }
    __syncthreads();
  }

  const int co = n0 + n;
  float sc = scale ? scale[co] : 1.f;
  float sh = shift ? shift[co] : 0.f;
#pragma unroll
  for (int r = 0; r < 8; ++r) {
    int mm = r + (kh << 3);
    int gx = x0 + wv * 16 + mm;
    if (gx < W) {
      float v = acc[r] * sc + sh;
      if (relu) v = fmaxf(v, 0.f);
      out[(((long)b * C + co) * H + y) * W + gx] = v;
    }
  }
}

// =====================================================================
// Generic bf16 WMMA GEMM:  C = relu?( A[M,K] @ W[N,K]^T + bias + add )
// W is bf16 in natural [N,K] (torch [out,in]) layout -> B-fragments are
// direct 16B global vector loads.  Block 128 thr: 64 rows x 16 cols.
// grid: (ceil(M/64), N/16, batches)
// =====================================================================
__global__ void gemm_bf16_wmma(const __bf16* __restrict__ A, long strideA, int lda,
                               const __bf16* __restrict__ Wt,   // [N, Kd]
                               const float* __restrict__ bias,
                               const float* __restrict__ add,
                               long add_bs, long add_rs, long add_cs,
                               float* __restrict__ Cf, long strideCf, int ldcf,
                               __bf16* __restrict__ Cb, long strideCb, int ldcb,
                               int M, int N, int Kd, int relu)
{
  __shared__ __align__(16) __bf16 As[64][40];   // stride 40 keeps 16B alignment
  const int tid = threadIdx.x, lane = tid & 31, wv = tid >> 5;
  const int m0 = blockIdx.x * 64;
  const int n0 = blockIdx.y * 16;
  const int batch = blockIdx.z;
  const __bf16* Ab = A + (long)batch * strideA;

  v8f acc;
#pragma unroll
  for (int i = 0; i < 8; ++i) acc[i] = 0.f;
  const int m = lane & 15, kh = lane >> 4, n = lane & 15;
  const __bf16* wrow = Wt + (long)(n0 + n) * Kd;

  for (int kb = 0; kb < Kd; kb += 32) {
    // stage A tile as 16B chunks: 64 rows x 4 chunks = 256 chunks / 128 thr
#pragma unroll
    for (int t = 0; t < 2; ++t) {
      int j = tid + t * 128;
      int r = j >> 2, cs = (j & 3) << 3;
      int gm = m0 + r;
      v8bf val = zero8();
      if (gm < M) val = ld8(&Ab[(long)gm * lda + kb + cs]);
      st8(&As[r][cs], val);
    }
    __syncthreads();
    const __bf16* ar = &As[wv * 16 + m][0];
    v16bf a  = cat16(ld8(ar + kh * 16), ld8(ar + kh * 16 + 8));
    const __bf16* br = wrow + kb + kh * 16;
    v16bf bf = cat16(ld8(br), ld8(br + 8));
    acc = __builtin_amdgcn_wmma_f32_16x16x32_bf16(false, a, false, bf,
                                                  (short)0, acc, false, false);
    __syncthreads();
  }

  const int gn = n0 + n;
  const float bz = bias ? bias[gn] : 0.f;
#pragma unroll
  for (int r = 0; r < 8; ++r) {
    int gm = m0 + wv * 16 + r + (kh << 3);
    if (gm < M) {
      float v = acc[r] + bz;
      if (add) v += add[(long)batch * add_bs + (long)gm * add_rs + (long)gn * add_cs];
      if (relu) v = fmaxf(v, 0.f);
      if (Cf) Cf[(long)batch * strideCf + (long)gm * ldcf + gn] = v;
      if (Cb) Cb[(long)batch * strideCb + (long)gm * ldcb + gn] = tobf(v);
    }
  }
}

// =====================================================================
// Flash-style attention, one wave per (q-tile, head, batch).
// qp/kp/vp: bf16 [B, Mq|Nk, C];  out: f32 (+ optional bf16) [B, Mq, C]
// =====================================================================
__global__ void flash_attn(const __bf16* __restrict__ qp,
                           const __bf16* __restrict__ kp,
                           const __bf16* __restrict__ vp,
                           float* __restrict__ outf, __bf16* __restrict__ outb,
                           int Mq, int Nk)
{
  __shared__ __align__(16) __bf16 lp[16][32];    // P tile  [q][key]
  __shared__ __align__(16) __bf16 lvt[16][32];   // V tile  [dh][key]
  const int lane = threadIdx.x;
  const int qt = blockIdx.x, h = blockIdx.y, b = blockIdx.z;
  const int m = lane & 15, kh = lane >> 4, n = lane & 15;
  const int qrow = qt * 16 + m;
  const float scl = 0.25f;              // 1/sqrt(DH)
  const v8bf z8 = zero8();

  // Q fragment: real data only at k<16 -> khalf picks its 8-elem chunk,
  // the k>=16 zero-pad chunk stays a register zero.
  v16bf aq;
  {
    v8bf lo = z8;
    if (qrow < Mq) lo = ld8(&qp[((long)b * Mq + qrow) * C + h * DH + kh * 8]);
    aq = cat16(lo, z8);
  }

  float mrun[8], lrun[8];
  v8f acc;
#pragma unroll
  for (int r = 0; r < 8; ++r) { mrun[r] = -3.0e38f; lrun[r] = 0.f; acc[r] = 0.f; }

  for (int kb = 0; kb < Nk; kb += 32) {
    // stage V tile transposed: lane owns key kb+lane, 16 dh values (2x16B loads)
    {
      int key = kb + lane;
      v8bf w0 = z8, w1 = z8;
      if (key < Nk) {
        const __bf16* vr = &vp[((long)b * Nk + key) * C + h * DH];
        w0 = ld8(vr); w1 = ld8(vr + 8);
      }
#pragma unroll
      for (int d = 0; d < 8; ++d)  lvt[d][lane]     = w0[d];
#pragma unroll
      for (int d = 0; d < 8; ++d)  lvt[8 + d][lane] = w1[d];
    }

    float s[2][8];
#pragma unroll
    for (int sub = 0; sub < 2; ++sub) {
      int key = kb + sub * 16 + n;
      // K^T fragment: contraction over dh; khalf picks its 8-elem chunk, k>=16 chunk is the zero pad.
      v16bf bk;
      {
        v8bf l0 = z8, l1 = z8;
        if (key < Nk) {
          const __bf16* kr = &kp[((long)b * Nk + key) * C + h * DH];
          l0 = ld8(kr + kh * 8);
        }
        bk = cat16(l0, l1);
      }
      v8f sc;
#pragma unroll
      for (int r = 0; r < 8; ++r) sc[r] = 0.f;
      sc = __builtin_amdgcn_wmma_f32_16x16x32_bf16(false, aq, false, bk,
                                                   (short)0, sc, false, false);
#pragma unroll
      for (int r = 0; r < 8; ++r) {
        float v = sc[r] * scl;
        if (key >= Nk) v = -1.0e30f;
        s[sub][r] = v;
      }
    }
    // streaming softmax; row m = r + 8*khalf lives across a 16-lane half
#pragma unroll
    for (int r = 0; r < 8; ++r) {
      float t = fmaxf(s[0][r], s[1][r]);
#pragma unroll
      for (int off = 1; off < 16; off <<= 1) t = fmaxf(t, __shfl_xor(t, off, 16));
      float mnew = fmaxf(mrun[r], t);
      float p0 = __expf(s[0][r] - mnew);
      float p1 = __expf(s[1][r] - mnew);
      float rs = p0 + p1;
#pragma unroll
      for (int off = 1; off < 16; off <<= 1) rs += __shfl_xor(rs, off, 16);
      float alpha = __expf(mrun[r] - mnew);
      lrun[r] = lrun[r] * alpha + rs;
      acc[r] *= alpha;
      mrun[r] = mnew;
      int row = r + (kh << 3);
      lp[row][n]      = tobf(p0);
      lp[row][16 + n] = tobf(p1);
    }
    __syncthreads();
    // P (A-frag, 2x16B) and V (B-frag, 2x16B) from LDS
    const __bf16* pr = &lp[m][0];
    v16bf ap = cat16(ld8(pr + kh * 16), ld8(pr + kh * 16 + 8));
    const __bf16* vr = &lvt[n][kh * 16];
    v16bf bv = cat16(ld8(vr), ld8(vr + 8));
    acc = __builtin_amdgcn_wmma_f32_16x16x32_bf16(false, ap, false, bv,
                                                  (short)0, acc, false, false);
    __syncthreads();
  }

#pragma unroll
  for (int r = 0; r < 8; ++r) {
    int row = qt * 16 + r + (kh << 3);
    if (row < Mq) {
      float v = acc[r] / lrun[r];
      long o = ((long)b * Mq + row) * C + h * DH + n;
      outf[o] = v;
      if (outb) outb[o] = tobf(v);
    }
  }
}

// =====================================================================
// 1x1 heatmap conv + sigmoid
// =====================================================================
__global__ void hm2_sigmoid(const float* __restrict__ hin, const float* __restrict__ w,
                            const float* __restrict__ bias, float* __restrict__ hm)
{
  long idx = (long)blockIdx.x * 256 + threadIdx.x;
  if (idx >= (long)Bsz * NCLS * HW) return;
  int pix = idx % HW; long t = idx / HW; int cls = t % NCLS; int b = (int)(t / NCLS);
  float s = bias[cls];
  for (int c = 0; c < C; ++c)
    s += hin[((long)b * C + c) * HW + pix] * w[cls * C + c];
  hm[idx] = 1.f / (1.f + __expf(-s));
}

// 3x3 local-max NMS masking (classes 8,9 pass through)
__global__ void nms_kernel(const float* __restrict__ hm, float* __restrict__ hmm)
{
  long i = (long)blockIdx.x * 256 + threadIdx.x;
  if (i >= (long)Bsz * NCLS * HW) return;
  int pix = i % HW; long t = i / HW; int cls = t % NCLS;
  int y = pix / W, x = pix % W;
  float v = hm[i];
  float lm = 0.f;
  if (y >= 1 && y < H - 1 && x >= 1 && x < W - 1) {
    lm = -3.0e38f;
    for (int dy = -1; dy <= 1; ++dy)
      for (int dx = -1; dx <= 1; ++dx)
        lm = fmaxf(lm, hm[i + dy * W + dx]);
  }
  if (cls == 8 || cls == 9) lm = v;
  hmm[i] = (v == lm) ? v : 0.f;
}

// iterative top-K (first-occurrence tie-break), one block per batch
__global__ void topk_kernel(const float* __restrict__ hmm, float* __restrict__ scratch,
                            int* __restrict__ top_cls, int* __restrict__ top_idx)
{
  const int b = blockIdx.x, tid = threadIdx.x, NT = 256;
  const long Ntot = (long)NCLS * HW;
  const float* src = hmm + (long)b * Ntot;
  float* buf = scratch + (long)b * Ntot;
  for (long i = tid; i < Ntot; i += NT) buf[i] = src[i];
  __syncthreads();
  __shared__ float sv[256];
  __shared__ int   si[256];
  for (int k = 0; k < Kq; ++k) {
    float bv = -3.0e38f; int bi = 0x7fffffff;
    for (long i = tid; i < Ntot; i += NT) {
      float v = buf[i];
      if (v > bv || (v == bv && (int)i < bi)) { bv = v; bi = (int)i; }
    }
    sv[tid] = bv; si[tid] = bi;
    __syncthreads();
    for (int s = 128; s > 0; s >>= 1) {
      if (tid < s) {
        if (sv[tid + s] > sv[tid] ||
            (sv[tid + s] == sv[tid] && si[tid + s] < si[tid])) {
          sv[tid] = sv[tid + s]; si[tid] = si[tid + s];
        }
      }
      __syncthreads();
    }
    if (tid == 0) {
      int idx = si[0];
      top_cls[b * Kq + k] = idx / HW;
      top_idx[b * Kq + k] = idx % HW;
      buf[idx] = -3.0e38f;
    }
    __syncthreads();
  }
}

// gather query features + class encoding + learned self pos-embed
__global__ void build_queries(const float* __restrict__ feat,
                              const int* __restrict__ top_cls, const int* __restrict__ top_idx,
                              const float* __restrict__ w_ce, const float* __restrict__ b_ce,
                              const float* __restrict__ pw1, const float* __restrict__ pb1,
                              const float* __restrict__ pg,  const float* __restrict__ pb,
                              const float* __restrict__ pw2, const float* __restrict__ pb2,
                              float* __restrict__ q_f32, float* __restrict__ qpe_f32,
                              __bf16* __restrict__ t_bf, float* __restrict__ qpos)
{
  int k = blockIdx.x, b = blockIdx.y, c = threadIdx.x;   // 128 threads
  int pix = top_idx[b * Kq + k], cls = top_cls[b * Kq + k];
  float px = (float)(pix % W) + 0.5f;
  float py = (float)(pix / W) + 0.5f;
  float qv = feat[((long)b * C + c) * HW + pix] + w_ce[c * NCLS + cls] + b_ce[c];
  __shared__ float hsh[C];
  float hc = pw1[c * 2 + 0] * px + pw1[c * 2 + 1] * py + pb1[c];
  hc = fmaxf(hc * pg[c] + pb[c], 0.f);
  hsh[c] = hc;
  __syncthreads();
  float acc = pb2[c];
  for (int j = 0; j < C; ++j) acc += pw2[c * C + j] * hsh[j];
  long o = ((long)b * Kq + k) * C + c;
  q_f32[o] = qv; qpe_f32[o] = acc; t_bf[o] = tobf(qv + acc);
  if (c == 0) { qpos[(b * Kq + k) * 2 + 0] = px; qpos[(b * Kq + k) * 2 + 1] = py; }
}

// cross pos-embed layer 1 over the whole BEV grid (bf16 out for GEMM)
__global__ void crosspe_h(const float* __restrict__ w1, const float* __restrict__ b1,
                          const float* __restrict__ g,  const float* __restrict__ bb,
                          __bf16* __restrict__ hmid)
{
  long idx = (long)blockIdx.x * 256 + threadIdx.x;
  if (idx >= (long)Bsz * HW * C) return;
  int c = idx % C; long t = idx / C; int pix = t % HW;
  float px = (float)(pix % W) + 0.5f;
  float py = (float)(pix / W) + 0.5f;
  float h = w1[c * 2 + 0] * px + w1[c * 2 + 1] * py + b1[c];
  hmid[idx] = tobf(fmaxf(h * g[c] + bb[c], 0.f));
}

// weight prep: plain f32 -> bf16 cast, layout preserved ([N,K])
__global__ void cast_to_bf16(const float* __restrict__ src, __bf16* __restrict__ dst,
                             long n)
{
  long i = (long)blockIdx.x * 256 + threadIdx.x;
  if (i < n) dst[i] = tobf(src[i]);
}

// LayerNorm over C, optional extra fp32 addend only on the bf16 copy
__global__ void layernorm_kernel(const float* __restrict__ x,
                                 const float* __restrict__ g, const float* __restrict__ bvec,
                                 const float* __restrict__ addb,
                                 float* __restrict__ yf, __bf16* __restrict__ yb)
{
  int row = blockIdx.x, c = threadIdx.x;   // 128 threads
  __shared__ float red[C];
  float v = x[(long)row * C + c];
  red[c] = v; __syncthreads();
  for (int s = 64; s > 0; s >>= 1) { if (c < s) red[c] += red[c + s]; __syncthreads(); }
  float mean = red[0] / C; __syncthreads();
  float d = v - mean;
  red[c] = d * d; __syncthreads();
  for (int s = 64; s > 0; s >>= 1) { if (c < s) red[c] += red[c + s]; __syncthreads(); }
  float var = red[0] / C;
  float y = d * rsqrtf(var + 1e-5f) * g[c] + bvec[c];
  long o = (long)row * C + c;
  if (yf) yf[o] = y;
  if (yb) yb[o] = tobf(addb ? y + addb[o] : y);
}

// FFN prediction head:  Conv1d->BN->ReLU->Conv1d, write [B,30,Kq] slice
__global__ void head_kernel(const float* __restrict__ q, const float* __restrict__ qpos,
                            const float* __restrict__ w1, const float* __restrict__ g,
                            const float* __restrict__ bb, const float* __restrict__ w2,
                            const float* __restrict__ b2,
                            int od, int ch_off, int add_pos, float* __restrict__ out)
{
  int k = blockIdx.x, b = blockIdx.y, tid = threadIdx.x;  // 128 threads
  __shared__ float qs[C];
  __shared__ float hh[64];
  qs[tid] = q[((long)b * Kq + k) * C + tid];
  __syncthreads();
  if (tid < 64) {
    float a = 0.f;
    for (int c = 0; c < C; ++c) a += w1[tid * C + c] * qs[c];
    hh[tid] = fmaxf(a * g[tid] + bb[tid], 0.f);
  }
  __syncthreads();
  if (tid < od) {
    float a = b2[tid];
    for (int j = 0; j < 64; ++j) a += w2[tid * 64 + j] * hh[j];
    if (add_pos) a += qpos[((long)b * Kq + k) * 2 + tid];
    out[((long)b * 30 + ch_off + tid) * Kq + k] = a;
  }
}

// query heatmap score gather -> channels 20..29
__global__ void qhs_kernel(const float* __restrict__ hmm, const int* __restrict__ top_idx,
                           float* __restrict__ out)
{
  int i = blockIdx.x * 256 + threadIdx.x;
  if (i >= Bsz * NCLS * Kq) return;
  int k = i % Kq; int t = i / Kq; int cls = t % NCLS; int b = t / NCLS;
  out[((long)b * 30 + 20 + cls) * Kq + k] =
      hmm[((long)b * NCLS + cls) * HW + top_idx[b * Kq + k]];
}

// =====================================================================
// host launcher
// =====================================================================
extern "C" void kernel_launch(void* const* d_in, const int* in_sizes, int n_in,
                              void* d_out, int out_size, void* d_ws, size_t ws_size,
                              hipStream_t stream)
{
  auto F = [&](int i) { return (const float*)d_in[i]; };
  // input indices (setup_inputs insertion order, nested dicts depth-first)
  const float* X       = F(0);
  const float* Wsh     = F(1);  const float* Bsh   = F(2);
  const float* Whm1    = F(3);  const float* HmG   = F(4);  const float* HmB = F(5);
  const float* Whm2    = F(6);  const float* Bhm2  = F(7);
  const float* Wce     = F(8);  const float* Bce   = F(9);
  const float* SPE_w1  = F(10); const float* SPE_b1 = F(11);
  const float* SPE_g   = F(12); const float* SPE_b  = F(13);
  const float* SPE_w2  = F(14); const float* SPE_b2 = F(15);
  const float* CPE_w1  = F(16); const float* CPE_b1 = F(17);
  const float* CPE_g   = F(18); const float* CPE_b  = F(19);
  const float* CPE_w2  = F(20); const float* CPE_b2 = F(21);
  const float* SA_win  = F(22); const float* SA_bin = F(23);
  const float* SA_wout = F(24); const float* SA_bout = F(25);
  const float* CA_win  = F(26); const float* CA_bin = F(27);
  const float* CA_wout = F(28); const float* CA_bout = F(29);
  const float* FF_w1   = F(30); const float* FF_b1  = F(31);
  const float* FF_w2   = F(32); const float* FF_b2  = F(33);
  const float* LN1g = F(34); const float* LN1b = F(35);
  const float* LN2g = F(36); const float* LN2b = F(37);
  const float* LN3g = F(38); const float* LN3b = F(39);
  const int HEAD0 = 40;   // per head: w1, g, b, w2, b2

  // ---------------- workspace carving ----------------
  char* base = (char*)d_ws;
  size_t off = 0;
  auto alloc = [&](size_t bytes) -> char* {
    char* p = base + off;
    off += (bytes + 255) & ~(size_t)255;
    return p;
  };
  float*  feat   = (float*)alloc((size_t)Bsz * C * HW * 4);
  float*  hconv  = (float*)alloc((size_t)Bsz * C * HW * 4);
  float*  hm     = (float*)alloc((size_t)Bsz * NCLS * HW * 4);
  float*  hmm    = (float*)alloc((size_t)Bsz * NCLS * HW * 4);
  float*  tscr   = (float*)alloc((size_t)Bsz * NCLS * HW * 4);
  int*    top_cls = (int*)alloc((size_t)Bsz * Kq * 4);
  int*    top_idx = (int*)alloc((size_t)Bsz * Kq * 4);
  float*  q_f32  = (float*)alloc((size_t)Bsz * Kq * C * 4);
  float*  qpe    = (float*)alloc((size_t)Bsz * Kq * C * 4);
  float*  qpos   = (float*)alloc((size_t)Bsz * Kq * 2 * 4);
  __bf16* t_bf   = (__bf16*)alloc((size_t)Bsz * Kq * C * 2);
  __bf16* t2_bf  = (__bf16*)alloc((size_t)Bsz * Kq * C * 2);
  float*  preln  = (float*)alloc((size_t)Bsz * Kq * C * 4);
  float*  attnf  = (float*)alloc((size_t)Bsz * Kq * C * 4);
  __bf16* attnb  = (__bf16*)alloc((size_t)Bsz * Kq * C * 2);
  __bf16* qs_b   = (__bf16*)alloc((size_t)Bsz * Kq * C * 2);
  __bf16* ks_b   = (__bf16*)alloc((size_t)Bsz * Kq * C * 2);
  __bf16* vs_b   = (__bf16*)alloc((size_t)Bsz * Kq * C * 2);
  __bf16* qpc    = (__bf16*)alloc((size_t)Bsz * Kq * C * 2);
  __bf16* h1bf   = (__bf16*)alloc((size_t)Bsz * Kq * FFN * 2);
  __bf16* hmid   = (__bf16*)alloc((size_t)Bsz * HW * C * 2);
  __bf16* kk     = (__bf16*)alloc((size_t)Bsz * HW * C * 2);
  __bf16* kproj  = (__bf16*)alloc((size_t)Bsz * HW * C * 2);
  __bf16* vproj  = (__bf16*)alloc((size_t)Bsz * HW * C * 2);
  __bf16* cpeW2  = (__bf16*)alloc((size_t)C * C * 2);
  __bf16* sWq    = (__bf16*)alloc((size_t)C * C * 2);
  __bf16* sWk    = (__bf16*)alloc((size_t)C * C * 2);
  __bf16* sWv    = (__bf16*)alloc((size_t)C * C * 2);
  __bf16* sWo    = (__bf16*)alloc((size_t)C * C * 2);
  __bf16* cWq    = (__bf16*)alloc((size_t)C * C * 2);
  __bf16* cWk    = (__bf16*)alloc((size_t)C * C * 2);
  __bf16* cWv    = (__bf16*)alloc((size_t)C * C * 2);
  __bf16* cWo    = (__bf16*)alloc((size_t)C * C * 2);
  __bf16* f1W    = (__bf16*)alloc((size_t)FFN * C * 2);
  __bf16* f2W    = (__bf16*)alloc((size_t)C * FFN * 2);
  float*  outp   = (float*)d_out;

  const long QC = (long)Kq * C;           // per-batch q stride
  const long HC = (long)HW * C;           // per-batch kv stride

  // ---------------- 1) backbone convs (WMMA) ----------------
  dim3 cgrid((W + 63) / 64, H, Bsz * (C / 16));
  conv3x3_wmma<<<cgrid, 128, 0, stream>>>(X, Wsh, nullptr, Bsh, CIN, 0, feat);
  conv3x3_wmma<<<cgrid, 128, 0, stream>>>(feat, Whm1, HmG, HmB, C, 1, hconv);

  long nhm = (long)Bsz * NCLS * HW;
  hm2_sigmoid<<<(nhm + 255) / 256, 256, 0, stream>>>(hconv, Whm2, Bhm2, hm);
  nms_kernel<<<(nhm + 255) / 256, 256, 0, stream>>>(hm, hmm);
  topk_kernel<<<Bsz, 256, 0, stream>>>(hmm, tscr, top_cls, top_idx);

  // ---------------- 2) queries + pos embeds ----------------
  build_queries<<<dim3(Kq, Bsz), C, 0, stream>>>(feat, top_cls, top_idx, Wce, Bce,
      SPE_w1, SPE_b1, SPE_g, SPE_b, SPE_w2, SPE_b2, q_f32, qpe, t_bf, qpos);
  long npe = (long)Bsz * HW * C;
  crosspe_h<<<(npe + 255) / 256, 256, 0, stream>>>(CPE_w1, CPE_b1, CPE_g, CPE_b, hmid);

  // ---------------- 3) weight preps (f32 -> bf16, [N,K] kept) ----------------
  auto prep = [&](const float* src, __bf16* dst, long n) {
    cast_to_bf16<<<(n + 255) / 256, 256, 0, stream>>>(src, dst, n);
  };
  prep(CPE_w2, cpeW2, (long)C * C);
  prep(SA_win + 0 * C * C, sWq, (long)C * C);
  prep(SA_win + 1 * C * C, sWk, (long)C * C);
  prep(SA_win + 2 * C * C, sWv, (long)C * C);
  prep(SA_wout, sWo, (long)C * C);
  prep(CA_win + 0 * C * C, cWq, (long)C * C);
  prep(CA_win + 1 * C * C, cWk, (long)C * C);
  prep(CA_win + 2 * C * C, cWv, (long)C * C);
  prep(CA_wout, cWo, (long)C * C);
  prep(FF_w1, f1W, (long)FFN * C);
  prep(FF_w2, f2W, (long)C * FFN);

  // GEMM launch helper
  auto gemm = [&](const __bf16* A, long sA, int lda, const __bf16* Wt,
                  const float* bias, const float* add, long abs_, long ars, long acs,
                  float* Cf, long sCf, int ldcf, __bf16* Cb, long sCb, int ldcb,
                  int M, int N, int Kd, int relu) {
    dim3 g((M + 63) / 64, N / 16, Bsz);
    gemm_bf16_wmma<<<g, 128, 0, stream>>>(A, sA, lda, Wt, bias,
        add, abs_, ars, acs, Cf, sCf, ldcf, Cb, sCb, ldcb, M, N, Kd, relu);
  };

  // ---------------- 4) self attention ----------------
  gemm(t_bf, QC, C, sWq, SA_bin + 0 * C, nullptr, 0, 0, 0,
       nullptr, 0, 0, qs_b, QC, C, Kq, C, C, 0);
  gemm(t_bf, QC, C, sWk, SA_bin + 1 * C, nullptr, 0, 0, 0,
       nullptr, 0, 0, ks_b, QC, C, Kq, C, C, 0);
  gemm(t_bf, QC, C, sWv, SA_bin + 2 * C, nullptr, 0, 0, 0,
       nullptr, 0, 0, vs_b, QC, C, Kq, C, C, 0);
  flash_attn<<<dim3((Kq + 15) / 16, NH, Bsz), 32, 0, stream>>>(
      qs_b, ks_b, vs_b, attnf, attnb, Kq, Kq);
  gemm(attnb, QC, C, sWo, SA_bout, q_f32, QC, C, 1,
       preln, QC, C, nullptr, 0, 0, Kq, C, C, 0);
  layernorm_kernel<<<Bsz * Kq, C, 0, stream>>>(preln, LN1g, LN1b, qpe, q_f32, t2_bf);

  // ---------------- 5) cross attention ----------------
  gemm(hmid, HC, C, cpeW2, CPE_b2, feat, (long)C * HW, 1, HW,
       nullptr, 0, 0, kk, HC, C, HW, C, C, 0);
  gemm(kk, HC, C, cWk, CA_bin + 1 * C, nullptr, 0, 0, 0,
       nullptr, 0, 0, kproj, HC, C, HW, C, C, 0);
  gemm(kk, HC, C, cWv, CA_bin + 2 * C, nullptr, 0, 0, 0,
       nullptr, 0, 0, vproj, HC, C, HW, C, C, 0);
  gemm(t2_bf, QC, C, cWq, CA_bin + 0 * C, nullptr, 0, 0, 0,
       nullptr, 0, 0, qpc, QC, C, Kq, C, C, 0);
  flash_attn<<<dim3((Kq + 15) / 16, NH, Bsz), 32, 0, stream>>>(
      qpc, kproj, vproj, attnf, attnb, Kq, HW);
  gemm(attnb, QC, C, cWo, CA_bout, q_f32, QC, C, 1,
       preln, QC, C, nullptr, 0, 0, Kq, C, C, 0);
  layernorm_kernel<<<Bsz * Kq, C, 0, stream>>>(preln, LN2g, LN2b, nullptr, q_f32, t2_bf);

  // ---------------- 6) FFN ----------------
  gemm(t2_bf, QC, C, f1W, FF_b1, nullptr, 0, 0, 0,
       nullptr, 0, 0, h1bf, (long)Kq * FFN, FFN, Kq, FFN, C, 1);
  gemm(h1bf, (long)Kq * FFN, FFN, f2W, FF_b2, q_f32, QC, C, 1,
       preln, QC, C, nullptr, 0, 0, Kq, C, FFN, 0);
  layernorm_kernel<<<Bsz * Kq, C, 0, stream>>>(preln, LN3g, LN3b, nullptr, q_f32, nullptr);

  // ---------------- 7) prediction heads ----------------
  const int hod[6]  = {2, 1, 3, 2, 2, NCLS};
  const int hoff[6] = {0, 2, 3, 6, 8, 10};
  for (int hI = 0; hI < 6; ++hI) {
    int p = HEAD0 + hI * 5;
    head_kernel<<<dim3(Kq, Bsz), C, 0, stream>>>(
        q_f32, qpos, F(p + 0), F(p + 1), F(p + 2), F(p + 3), F(p + 4),
        hod[hI], hoff[hI], hI == 0 ? 1 : 0, outp);
  }
  qhs_kernel<<<(Bsz * NCLS * Kq + 255) / 256, 256, 0, stream>>>(hmm, top_idx, outp);
  (void)in_sizes; (void)n_in; (void)out_size; (void)ws_size;
}
